// BahdahnauAttentiveDecoder_52115133169751
// MI455X (gfx1250) — matrix operand order repaired and checked by reading, not compiled, not run on hardware
//
#include <hip/hip_runtime.h>
#include <hip/hip_bf16.h>

// ---------------------------------------------------------------------------
// CDNA5 WMMA types / helpers (gfx1250, wave32)
// ---------------------------------------------------------------------------
typedef __attribute__((ext_vector_type(16))) __bf16 bf16x16;
typedef __attribute__((ext_vector_type(8)))  float  floatx8;

#define WMMA_BF16(a, b, c) \
  __builtin_amdgcn_wmma_f32_16x16x32_bf16(false, (a), false, (b), (short)0, (c), false, false)

static __device__ __forceinline__ __bf16 us_to_bf(unsigned short u) {
  union { unsigned short s; __bf16 b; } cv; cv.s = u; return cv.b;
}
static __device__ __forceinline__ unsigned short f_to_us(float f) {
  unsigned int u = __builtin_bit_cast(unsigned int, f);
  unsigned int r = (u + 0x7FFFu + ((u >> 16) & 1u)) >> 16;  // RNE
  return (unsigned short)r;
}

// A-fragment (16x32 bf16, ISA 7.12.2 layout) built from fp32 row-major X.
// lane&15 -> M row; lane>>4 selects K half (offset 8); VGPR g holds K pair
// (g<4?0:16) + (g&3)*2 (+1).
static __device__ __forceinline__ bf16x16
load_a_f32(const float* __restrict__ X, int ldx, int mbase, int kc, int lane) {
  const float* row = X + (size_t)(mbase + (lane & 15)) * ldx + kc + ((lane >> 4) << 3);
  bf16x16 a;
#pragma unroll
  for (int g = 0; g < 8; ++g) {
    int kb = ((g & 4) << 2) + ((g & 3) << 1);
    float2 f = *(const float2*)(row + kb);
    a[2 * g]     = (__bf16)f.x;
    a[2 * g + 1] = (__bf16)f.y;
  }
  return a;
}

// Same, but rows gathered through an embedding index column.
static __device__ __forceinline__ bf16x16
load_a_emb(const float* __restrict__ emb, const int* __restrict__ ids, int ids_stride,
           int E, int mbase, int kc, int lane) {
  int m = mbase + (lane & 15);
  long long id = ids[(size_t)m * ids_stride];
  const float* row = emb + (size_t)id * E + kc + ((lane >> 4) << 3);
  bf16x16 a;
#pragma unroll
  for (int g = 0; g < 8; ++g) {
    int kb = ((g & 4) << 2) + ((g & 3) << 1);
    float2 f = *(const float2*)(row + kb);
    a[2 * g]     = (__bf16)f.x;
    a[2 * g + 1] = (__bf16)f.y;
  }
  return a;
}

// B-fragment (32x16 bf16) from bf16 weights stored [n][k] row-major.
// N = lane&15 ; K = (lane>>4)*8 + (v<4?0:16) + (v&3)*2 (+1).
static __device__ __forceinline__ bf16x16
load_b_us(const unsigned short* __restrict__ W, int ldw, int nbase, int kc, int lane) {
  const unsigned short* row = W + (size_t)(nbase + (lane & 15)) * ldw + kc + ((lane >> 4) << 3);
  bf16x16 b;
#pragma unroll
  for (int v = 0; v < 8; ++v) {
    unsigned int d = *(const unsigned int*)(row + ((v & 4) << 2) + ((v & 3) << 1));
    b[2 * v]     = us_to_bf((unsigned short)(d & 0xFFFFu));
    b[2 * v + 1] = us_to_bf((unsigned short)(d >> 16));
  }
  return b;
}

// C/D tile store: M = mbase + 8*(lane>>4) + v ; N = nbase + (lane&15).
static __device__ __forceinline__ void
store_tile(float* __restrict__ OUT, int ldo, int mbase, int nbase, int lane,
           floatx8 c, float bias) {
  int n = nbase + (lane & 15);
  int mt = (lane >> 4) << 3;
#pragma unroll
  for (int v = 0; v < 8; ++v)
    OUT[(size_t)(mbase + mt + v) * ldo + n] = c[v] + bias;
}

// ---------------------------------------------------------------------------
// Kernels
// ---------------------------------------------------------------------------
__global__ void f32_to_bf16_kernel(const float* __restrict__ src,
                                   unsigned short* __restrict__ dst, int n) {
  int i = blockIdx.x * blockDim.x + threadIdx.x;
  if (i < n) dst[i] = f_to_us(src[i]);
}

// Generic OUT[M,N] = X[M,K] * W[N,K]^T   (one 16x16 tile per wave)
__global__ void gemm_big_kernel(const float* __restrict__ X,
                                const unsigned short* __restrict__ W,
                                float* __restrict__ OUT, int K, int Ncols) {
  int lane = threadIdx.x;
  int nb = blockIdx.x * 16;
  int mb = blockIdx.y * 16;
  floatx8 c = {};
  for (int kc = 0; kc < K; kc += 32) {
    bf16x16 a = load_a_f32(X, K, mb, kc, lane);
    bf16x16 b = load_b_us(W, K, nb, kc, lane);
    c = WMMA_BF16(a, b, c);
  }
  store_tile(OUT, Ncols, mb, nb, lane, c, 0.f);
}

// M=32 specialized: two M-tiles per wave, B reused.
__global__ void gemm32_kernel(const float* __restrict__ X,
                              const unsigned short* __restrict__ W,
                              float* __restrict__ OUT, int K, int ldo) {
  int lane = threadIdx.x;
  int nb = blockIdx.x * 16;
  floatx8 c0 = {}, c1 = {};
  for (int kc = 0; kc < K; kc += 32) {
    bf16x16 b  = load_b_us(W, K, nb, kc, lane);
    bf16x16 a0 = load_a_f32(X, K, 0, kc, lane);
    bf16x16 a1 = load_a_f32(X, K, 16, kc, lane);
    c0 = WMMA_BF16(a0, b, c0);
    c1 = WMMA_BF16(a1, b, c1);
  }
  store_tile(OUT, ldo, 0,  nb, lane, c0, 0.f);
  store_tile(OUT, ldo, 16, nb, lane, c1, 0.f);
}

// Fused Bahdanau score + softmax + context. One block per batch row.
__global__ void attn_kernel(const float* __restrict__ enc_proj,   // [N][64][1024]
                            const float* __restrict__ dec_proj,   // [N][1024]
                            const float* __restrict__ v_att,      // [1024]
                            const float* __restrict__ enc_hids,   // [N][64][1024]
                            float* __restrict__ ctx,              // [N][1024]
                            float* __restrict__ att_out,          // base + t, stride att_ld
                            int att_ld) {
  __shared__ float sc[64];
  __shared__ float att[64];
  int n = blockIdx.x;
  int tid = threadIdx.x;
  int wave = tid >> 5, lane = tid & 31;
  const float* dp = dec_proj + (size_t)n * 1024;

  for (int l = wave; l < 64; l += 8) {
    const float* ep = enc_proj + ((size_t)n * 64 + l) * 1024;
    float s = 0.f;
    for (int a = lane; a < 1024; a += 32)
      s += tanhf(ep[a] + dp[a]) * v_att[a];
#pragma unroll
    for (int off = 16; off > 0; off >>= 1)
      s += __shfl_xor(s, off, 32);
    if (lane == 0) sc[l] = s;
  }
  __syncthreads();
  if (tid == 0) {
    float mx = sc[0];
    for (int l = 1; l < 64; ++l) mx = fmaxf(mx, sc[l]);
    float sum = 0.f;
    for (int l = 0; l < 64; ++l) { float e = __expf(sc[l] - mx); att[l] = e; sum += e; }
    float inv = 1.f / sum;
    for (int l = 0; l < 64; ++l) att[l] *= inv;
  }
  __syncthreads();
  if (tid < 64) att_out[((size_t)n * 64 + tid) * att_ld] = att[tid];
  for (int d = tid; d < 1024; d += blockDim.x) {
    float acc = 0.f;
    const float* eh = enc_hids + (size_t)n * 64 * 1024 + d;
#pragma unroll 4
    for (int l = 0; l < 64; ++l)
      acc += att[l] * eh[(size_t)l * 1024];
    ctx[(size_t)n * 1024 + d] = acc;
  }
}

// merged = emb[ids] @ W_in^T + b_in + ctx @ W_ctx^T + b_ctx   (M=32)
__global__ void merged_kernel(const int* __restrict__ ids_col, int ids_stride,
                              const float* __restrict__ emb,
                              const float* __restrict__ ctx,
                              const unsigned short* __restrict__ Win,
                              const unsigned short* __restrict__ Wctx,
                              const float* __restrict__ bin,
                              const float* __restrict__ bctx,
                              float* __restrict__ Xout) {
  int lane = threadIdx.x;
  int nb = blockIdx.x * 16;
  floatx8 c0 = {}, c1 = {};
  for (int kc = 0; kc < 1024; kc += 32) {
    bf16x16 bi = load_b_us(Win,  1024, nb, kc, lane);
    bf16x16 bc = load_b_us(Wctx, 1024, nb, kc, lane);
    bf16x16 a;
    a = load_a_emb(emb, ids_col, ids_stride, 1024, 0,  kc, lane); c0 = WMMA_BF16(a, bi, c0);
    a = load_a_emb(emb, ids_col, ids_stride, 1024, 16, kc, lane); c1 = WMMA_BF16(a, bi, c1);
    a = load_a_f32(ctx, 1024, 0,  kc, lane);                      c0 = WMMA_BF16(a, bc, c0);
    a = load_a_f32(ctx, 1024, 16, kc, lane);                      c1 = WMMA_BF16(a, bc, c1);
  }
  float bias = bin[nb + (lane & 15)] + bctx[nb + (lane & 15)];
  store_tile(Xout, 1024, 0,  nb, lane, c0, bias);
  store_tile(Xout, 1024, 16, nb, lane, c1, bias);
}

// One GRU layer: 6 WMMA accumulators (r/z/n x ih/hh) + fused gates.
// grid = (H/16 col tiles, 2 m-tiles), block = 32 (one wave).
__global__ void gru_kernel(const float* __restrict__ X,
                           const float* __restrict__ Hprev,
                           const unsigned short* __restrict__ Wih,  // [3H][H] bf16
                           const unsigned short* __restrict__ Whh,  // [3H][H] bf16
                           const float* __restrict__ bih,           // [3H]
                           const float* __restrict__ bhh,           // [3H]
                           float* __restrict__ Hnew,
                           float* __restrict__ dech, int dech_ld) {
  const int H = 1024;
  int lane = threadIdx.x;
  int nb = blockIdx.x * 16;
  int mb = blockIdx.y * 16;
  floatx8 cir = {}, ciz = {}, cin = {}, chr_ = {}, chz = {}, chn = {};
  for (int kc = 0; kc < H; kc += 32) {
    bf16x16 ax = load_a_f32(X,     H, mb, kc, lane);
    bf16x16 ah = load_a_f32(Hprev, H, mb, kc, lane);
    if (kc + 32 < H) {  // hint next K-chunk of the streamed weights (L2-resident)
      __builtin_prefetch(Wih + (size_t)(nb + (lane & 15)) * H + kc + 32, 0, 0);
      __builtin_prefetch(Whh + (size_t)(nb + (lane & 15)) * H + kc + 32, 0, 0);
    }
    bf16x16 b;
    b = load_b_us(Wih, H, 0 * H + nb, kc, lane); cir  = WMMA_BF16(ax, b, cir);
    b = load_b_us(Wih, H, 1 * H + nb, kc, lane); ciz  = WMMA_BF16(ax, b, ciz);
    b = load_b_us(Wih, H, 2 * H + nb, kc, lane); cin  = WMMA_BF16(ax, b, cin);
    b = load_b_us(Whh, H, 0 * H + nb, kc, lane); chr_ = WMMA_BF16(ah, b, chr_);
    b = load_b_us(Whh, H, 1 * H + nb, kc, lane); chz  = WMMA_BF16(ah, b, chz);
    b = load_b_us(Whh, H, 2 * H + nb, kc, lane); chn  = WMMA_BF16(ah, b, chn);
  }
  int n = nb + (lane & 15);
  int mt = (lane >> 4) << 3;
  float b_ir = bih[0 * H + n], b_iz = bih[1 * H + n], b_in_ = bih[2 * H + n];
  float b_hr = bhh[0 * H + n], b_hz = bhh[1 * H + n], b_hn  = bhh[2 * H + n];
#pragma unroll
  for (int v = 0; v < 8; ++v) {
    int m = mb + mt + v;
    float ir = cir[v]  + b_ir, iz = ciz[v] + b_iz, in_ = cin[v] + b_in_;
    float hr = chr_[v] + b_hr, hz = chz[v] + b_hz, hn  = chn[v] + b_hn;
    float r  = 1.f / (1.f + __expf(-(ir + hr)));
    float z  = 1.f / (1.f + __expf(-(iz + hz)));
    float nn = tanhf(in_ + r * hn);
    float hp = Hprev[(size_t)m * H + n];
    float ho = (1.f - z) * nn + z * hp;
    Hnew[(size_t)m * H + n] = ho;
    if (dech) dech[(size_t)m * dech_ld + n] = ho;
  }
}

// ---------------------------------------------------------------------------
// Host launch
// ---------------------------------------------------------------------------
extern "C" void kernel_launch(void* const* d_in, const int* in_sizes, int n_in,
                              void* d_out, int out_size, void* d_ws, size_t ws_size,
                              hipStream_t stream) {
  const int H = 1024, E = 1024, A = 1024, ENC_D = 1024;
  const int ENC_L = 64, N = 32, NL = 2, DEC_L = 64;
  const int T = DEC_L - 1;  // 63 steps

  const int*   input_ids = (const int*)  d_in[0];
  const float* dec_init  = (const float*)d_in[1];
  const float* enc_hids  = (const float*)d_in[2];
  const float* emb_table = (const float*)d_in[3];
  const float* W_in      = (const float*)d_in[4];
  const float* b_in      = (const float*)d_in[5];
  const float* W_ctx     = (const float*)d_in[6];
  const float* b_ctx     = (const float*)d_in[7];
  const float* W_att_dec = (const float*)d_in[8];
  const float* W_att_enc = (const float*)d_in[9];
  const float* v_att     = (const float*)d_in[10];
  const float* W_ih      = (const float*)d_in[11];
  const float* W_hh      = (const float*)d_in[12];
  const float* b_ih      = (const float*)d_in[13];
  const float* b_hh      = (const float*)d_in[14];
  (void)in_sizes; (void)n_in; (void)out_size; (void)ws_size;

  // Workspace carve-up
  char* w = (char*)d_ws;
  auto carve = [&](size_t bytes) -> void* {
    void* p = (void*)w;
    w += (bytes + 255) & ~(size_t)255;
    return p;
  };
  unsigned short* wbIn     = (unsigned short*)carve((size_t)H * E * 2);
  unsigned short* wbCtx    = (unsigned short*)carve((size_t)H * ENC_D * 2);
  unsigned short* wbAttDec = (unsigned short*)carve((size_t)A * H * 2);
  unsigned short* wbAttEnc = (unsigned short*)carve((size_t)A * ENC_D * 2);
  unsigned short* wbIh     = (unsigned short*)carve((size_t)NL * 3 * H * H * 2);
  unsigned short* wbHh     = (unsigned short*)carve((size_t)NL * 3 * H * H * 2);
  float* encp   = (float*)carve((size_t)N * ENC_L * A * 4);
  float* decp   = (float*)carve((size_t)N * A * 4);
  float* ctxbuf = (float*)carve((size_t)N * ENC_D * 4);
  float* xbuf   = (float*)carve((size_t)N * H * 4);
  float* hA     = (float*)carve((size_t)NL * N * H * 4);
  float* hB     = (float*)carve((size_t)NL * N * H * 4);

  // Weights -> bf16 (stay resident in 192MB L2 across all 63 steps)
  const int thr = 256;
  auto cvt = [&](const float* s, unsigned short* d, size_t n) {
    f32_to_bf16_kernel<<<dim3((unsigned)((n + thr - 1) / thr)), thr, 0, stream>>>(s, d, (int)n);
  };
  cvt(W_in,      wbIn,     (size_t)H * E);
  cvt(W_ctx,     wbCtx,    (size_t)H * ENC_D);
  cvt(W_att_dec, wbAttDec, (size_t)A * H);
  cvt(W_att_enc, wbAttEnc, (size_t)A * ENC_D);
  cvt(W_ih,      wbIh,     (size_t)NL * 3 * H * H);
  cvt(W_hh,      wbHh,     (size_t)NL * 3 * H * H);

  // enc_proj = enc_hids(2048,1024) @ W_att_enc^T   (once)
  gemm_big_kernel<<<dim3(A / 16, (N * ENC_L) / 16), 32, 0, stream>>>(
      enc_hids, wbAttEnc, encp, ENC_D, A);

  hipMemcpyAsync(hA, dec_init, (size_t)NL * N * H * sizeof(float),
                 hipMemcpyDeviceToDevice, stream);

  float* dec_hids_out = (float*)d_out;                       // [N][T][H]
  float* atts_out     = dec_hids_out + (size_t)N * T * H;    // [N][ENC_L][T]
  float* dec_prev_out = atts_out + (size_t)N * ENC_L * T;    // [NL][N][H]

  for (int t = 0; t < T; ++t) {
    float* h_in  = (t & 1) ? hB : hA;
    float* h_out = (t & 1) ? hA : hB;

    // dec_proj from top-layer hidden state
    gemm32_kernel<<<A / 16, 32, 0, stream>>>(
        h_in + (size_t)(NL - 1) * N * H, wbAttDec, decp, H, A);

    // score + softmax + context (writes atts[:, :, t])
    attn_kernel<<<N, 256, 0, stream>>>(encp, decp, v_att, enc_hids,
                                       ctxbuf, atts_out + t, T);

    // merged input (embedding gather fused into A-operand loads)
    merged_kernel<<<H / 16, 32, 0, stream>>>(
        input_ids + t, DEC_L, emb_table, ctxbuf, wbIn, wbCtx, b_in, b_ctx, xbuf);

    // GRU layer 0
    gru_kernel<<<dim3(H / 16, 2), 32, 0, stream>>>(
        xbuf, h_in, wbIh, wbHh, b_ih, b_hh, h_out, (float*)nullptr, 0);

    // GRU layer 1 (also streams dec_hids[:, t, :] to d_out)
    gru_kernel<<<dim3(H / 16, 2), 32, 0, stream>>>(
        h_out, h_in + (size_t)N * H,
        wbIh + (size_t)3 * H * H, wbHh + (size_t)3 * H * H,
        b_ih + 3 * H, b_hh + 3 * H,
        h_out + (size_t)N * H,
        dec_hids_out + (size_t)t * H, T * H);
  }

  float* h_fin = (T & 1) ? hB : hA;  // T=63 -> hB
  hipMemcpyAsync(dec_prev_out, h_fin, (size_t)NL * N * H * sizeof(float),
                 hipMemcpyDeviceToDevice, stream);
}
